// SS2D_48112223650166
// MI455X (gfx1250) — compile-verified, matrix operand order checked by reading, not run on hardware
//
#include <hip/hip_runtime.h>
#include <hip/hip_bf16.h>
#include <math.h>

#define B0   4
#define H0   64
#define W0   64
#define C0   128
#define DI   256
#define NST  16
#define RK   8
#define KDIR 4
#define L0   (H0*W0)      // 4096
#define M0   (B0*L0)      // 16384
#define NPROJ 64          // 40 padded to 64
#define EPSF 1e-5f

typedef __attribute__((ext_vector_type(16))) __bf16 v16bf;
typedef __attribute__((ext_vector_type(8)))  __bf16 v8bf;
typedef __attribute__((ext_vector_type(8)))  float  v8f;
typedef int v4i __attribute__((vector_size(16)));

#if defined(__has_builtin)
#if __has_builtin(__builtin_amdgcn_global_load_async_to_lds_b128) && \
    __has_builtin(__builtin_amdgcn_s_wait_asynccnt)
#define USE_ASYNC_LDS 1
#endif
#endif

// ---------------- elementwise prep kernels ----------------

__global__ void k_f32_to_bf16(const float* __restrict__ src,
                              __bf16* __restrict__ dst, int n) {
  int i = blockIdx.x * 256 + threadIdx.x;
  if (i < n) dst[i] = (__bf16)src[i];
}

__global__ void k_pack_proj(const float* __restrict__ w,   // (4, 40, 256)
                            __bf16* __restrict__ dst) {    // (4, 64, 256) zero-padded
  int i = blockIdx.x * 256 + threadIdx.x;
  if (i >= KDIR * NPROJ * DI) return;
  int d   = i % DI;
  int row = (i / DI) % NPROJ;
  int k   = i / (DI * NPROJ);
  float v = (row < (RK + 2 * NST)) ? w[(k * (RK + 2 * NST) + row) * DI + d] : 0.f;
  dst[i] = (__bf16)v;
}

__global__ void k_prep_A(const float* __restrict__ A_logs,
                         float* __restrict__ Aneg, int n) {
  int i = blockIdx.x * 256 + threadIdx.x;
  if (i < n) Aneg[i] = -expf(A_logs[i]);
}

__global__ void k_zero(float* __restrict__ p, int n) {
  int i = blockIdx.x * 256 + threadIdx.x;
  if (i < n) p[i] = 0.f;
}

// ---------------- WMMA bf16 GEMM: C[M,N] = A[M,K] * W[N,K]^T ----------------
// block = 256 threads = 8 waves; block tile = 128(M) x 64(N).
// wave w owns a 16x64 strip: 4 accumulators, A fragment reused across 4
// back-to-back v_wmma_f32_16x16x32_bf16 (XDL pipelining). A slab (128x32)
// staged in LDS per K-step, via async-load-to-LDS when the toolchain has it.

__global__ __launch_bounds__(256) void k_gemm_bf16(
    const __bf16* __restrict__ A, const __bf16* __restrict__ W,
    float* __restrict__ C, int M, int N, int K) {
  __shared__ __bf16 smA[128 * 40];              // stride 40 -> 16B-aligned chunks
  const int lane  = threadIdx.x & 31;
  const int wave  = threadIdx.x >> 5;           // 0..7 -> M sub-tile
  const int tileM = blockIdx.x * 128;
  const int tileN = blockIdx.y * 64;
  const int row16 = lane & 15;
  const int half  = lane >> 4;                  // 0: K 0..7,16..23  1: K 8..15,24..31
  const int koff  = half * 8;

  v8f acc[4] = {};

  const int srow  = threadIdx.x >> 1;           // 0..127
  const int scol0 = (threadIdx.x & 1) * 16;     // 0 or 16
  const __bf16* Astage = A + (size_t)(tileM + srow) * K;

  for (int kb = 0; kb < K; kb += 32) {
    __syncthreads();                            // WAR: prev fragment reads done
#ifdef USE_ASYNC_LDS
    __builtin_amdgcn_global_load_async_to_lds_b128(
        (__attribute__((address_space(1))) v4i*)(Astage + kb + scol0),
        (__attribute__((address_space(3))) v4i*)&smA[srow * 40 + scol0], 0, 0);
    __builtin_amdgcn_global_load_async_to_lds_b128(
        (__attribute__((address_space(1))) v4i*)(Astage + kb + scol0 + 8),
        (__attribute__((address_space(3))) v4i*)&smA[srow * 40 + scol0 + 8], 0, 0);
    __builtin_amdgcn_s_wait_asynccnt(0);
#else
    v8bf s0 = *(const v8bf*)(Astage + kb + scol0);
    v8bf s1 = *(const v8bf*)(Astage + kb + scol0 + 8);
    *(v8bf*)(&smA[srow * 40 + scol0])     = s0;
    *(v8bf*)(&smA[srow * 40 + scol0 + 8]) = s1;
#endif
    __syncthreads();
    if (kb + 32 < K) __builtin_prefetch(Astage + kb + 32 + scol0, 0, 0);

    const int ar = wave * 16 + row16;
    v8bf a0 = *(const v8bf*)(&smA[ar * 40 + koff]);
    v8bf a1 = *(const v8bf*)(&smA[ar * 40 + koff + 16]);
    v16bf af;
#pragma unroll
    for (int i = 0; i < 8; ++i) { af[i] = a0[i]; af[i + 8] = a1[i]; }

#pragma unroll
    for (int nt = 0; nt < 4; ++nt) {
      const __bf16* Wr = W + (size_t)(tileN + nt * 16 + row16) * K;
      v8bf b0 = *(const v8bf*)(Wr + kb + koff);
      v8bf b1 = *(const v8bf*)(Wr + kb + koff + 16);
      v16bf bf;
#pragma unroll
      for (int i = 0; i < 8; ++i) { bf[i] = b0[i]; bf[i + 8] = b1[i]; }
      acc[nt] = __builtin_amdgcn_wmma_f32_16x16x32_bf16(
          false, af, false, bf, (short)0, acc[nt], false, false);
    }
  }

  const int crow = tileM + wave * 16 + half * 8; // C/D: M split at 8 by lane half
#pragma unroll
  for (int nt = 0; nt < 4; ++nt) {
    const int ccol = tileN + nt * 16 + row16;
#pragma unroll
    for (int i = 0; i < 8; ++i)
      C[(size_t)(crow + i) * N + ccol] = acc[nt][i];
  }
}

// ---------------- depthwise 3x3 conv + bias + SiLU ----------------

__global__ __launch_bounds__(256) void k_conv_dw(
    const float* __restrict__ xz,        // (M0, 512), xc = cols 0..255
    const float* __restrict__ cw,        // (256, 9)
    const float* __restrict__ cb,        // (256)
    float*  __restrict__ xc_f32,         // (M0, 256)
    __bf16* __restrict__ xs0,            // (M0, 256) ordering 0
    __bf16* __restrict__ xs1) {          // (M0, 256) ordering 1 (HW transposed)
  int d = threadIdx.x;
  int l = blockIdx.x % L0;
  int b = blockIdx.x / L0;
  int h = l >> 6, w = l & 63;
  float acc = 0.f;
#pragma unroll
  for (int dy = -1; dy <= 1; ++dy) {
    int hh = h + dy;
    if (hh < 0 || hh >= H0) continue;
#pragma unroll
    for (int dx = -1; dx <= 1; ++dx) {
      int ww = w + dx;
      if (ww < 0 || ww >= W0) continue;
      float v = xz[((size_t)(b * L0 + hh * W0 + ww)) * 512 + d];
      acc = fmaf(v, cw[d * 9 + (dy + 1) * 3 + (dx + 1)], acc);
    }
  }
  acc += cb[d];
  float sv = acc / (1.f + expf(-acc));          // silu
  size_t i0 = ((size_t)(b * L0) + l) * DI + d;
  xc_f32[i0] = sv;
  xs0[i0] = (__bf16)sv;
  int m1 = w * H0 + h;
  xs1[((size_t)(b * L0) + m1) * DI + d] = (__bf16)sv;
}

// ---------------- selective scan ----------------

__global__ __launch_bounds__(256) void k_scan(
    const float* __restrict__ xc,     // (M0, 256) spatial NHWC
    const float* __restrict__ xdbl,   // (4, M0, 64): [k][(b*L+m)*64 + c]
    const float* __restrict__ dtw,    // (4, 256, 8)
    const float* __restrict__ dtb,    // (4, 256)
    const float* __restrict__ Aneg,   // (4*256, 16)  = -exp(A_logs)
    const float* __restrict__ Ds,     // (4*256)
    float* __restrict__ yo0,          // (M0, 256) ordering-0 accumulator
    float* __restrict__ yo1) {        // (M0, 256) ordering-1 accumulator
  __shared__ float sbuf[2][40];
  int d = threadIdx.x;
  int k = blockIdx.x & 3;
  int b = blockIdx.x >> 2;
  int o = k & 1;
  float* yo = o ? yo1 : yo0;

  float h[NST], Ar[NST], wr[RK];
#pragma unroll
  for (int n = 0; n < NST; ++n) {
    h[n]  = 0.f;
    Ar[n] = Aneg[((k * DI) + d) * NST + n];
  }
#pragma unroll
  for (int r = 0; r < RK; ++r) wr[r] = dtw[((k * DI) + d) * RK + r];
  float bias = dtb[k * DI + d];
  float Dv   = Ds[k * DI + d];
  const float* xdk = xdbl + (size_t)k * M0 * 64 + (size_t)b * L0 * 64;

  int mfirst = (k < 2) ? 0 : (L0 - 1);
  if (d < 40) sbuf[0][d] = xdk[(size_t)mfirst * 64 + d];
  __syncthreads();

  for (int t = 0; t < L0; ++t) {
    int m   = (k < 2) ? t : (L0 - 1 - t);
    int cur = t & 1;
    if (t + 1 < L0) {
      int mn = (k < 2) ? (t + 1) : (L0 - 2 - t);
      if (d < 40) sbuf[cur ^ 1][d] = xdk[(size_t)mn * 64 + d];
    }
    int l = o ? ((m & 63) * 64 + (m >> 6)) : m;   // ordering -> spatial
    float u = xc[((size_t)(b * L0) + l) * DI + d];

    float dt = bias;
#pragma unroll
    for (int r = 0; r < RK; ++r) dt = fmaf(wr[r], sbuf[cur][r], dt);
    dt = (dt > 20.f) ? dt : log1pf(expf(dt));     // softplus
    float du = dt * u;
    float y  = Dv * u;
#pragma unroll
    for (int n = 0; n < NST; ++n) {
      float e = __expf(dt * Ar[n]);
      h[n] = fmaf(e, h[n], du * sbuf[cur][8 + n]);
      y    = fmaf(h[n], sbuf[cur][24 + n], y);
    }
    atomicAdd(&yo[((size_t)(b * L0) + m) * DI + d], y);
    __syncthreads();
  }
}

// ---------------- cross-merge + LayerNorm + SiLU gate ----------------

__global__ __launch_bounds__(256) void k_merge_ln(
    const float* __restrict__ yo0, const float* __restrict__ yo1,
    const float* __restrict__ xz,                 // z = cols 256..511
    const float* __restrict__ lnw, const float* __restrict__ lnb,
    __bf16* __restrict__ ybf) {
  __shared__ float red[256];
  int d = threadIdx.x;
  int l = blockIdx.x % L0;
  int b = blockIdx.x / L0;
  int m1 = (l & 63) * 64 + (l >> 6);
  float yv = yo0[((size_t)(b * L0) + l) * DI + d] +
             yo1[((size_t)(b * L0) + m1) * DI + d];

  red[d] = yv; __syncthreads();
  for (int s = 128; s > 0; s >>= 1) { if (d < s) red[d] += red[d + s]; __syncthreads(); }
  float mu = red[0] / (float)DI;
  __syncthreads();
  float c = yv - mu;
  red[d] = c * c; __syncthreads();
  for (int s = 128; s > 0; s >>= 1) { if (d < s) red[d] += red[d + s]; __syncthreads(); }
  float var = red[0] / (float)DI;

  float yn = c * rsqrtf(var + EPSF) * lnw[d] + lnb[d];
  float z  = xz[((size_t)(b * L0) + l) * 512 + 256 + d];
  z = z / (1.f + expf(-z));                      // silu gate
  ybf[((size_t)(b * L0) + l) * DI + d] = (__bf16)(yn * z);
}

// ---------------- host orchestration ----------------

extern "C" void kernel_launch(void* const* d_in, const int* in_sizes, int n_in,
                              void* d_out, int out_size, void* d_ws, size_t ws_size,
                              hipStream_t stream) {
  (void)in_sizes; (void)n_in; (void)out_size; (void)ws_size;
  const float* xin[2]  = {(const float*)d_in[0],  (const float*)d_in[1]};
  const float* inw[2]  = {(const float*)d_in[2],  (const float*)d_in[3]};
  const float* cw[2]   = {(const float*)d_in[4],  (const float*)d_in[6]};
  const float* cb[2]   = {(const float*)d_in[5],  (const float*)d_in[7]};
  const float* xproj   = (const float*)d_in[8];
  const float* dtw     = (const float*)d_in[9];
  const float* dtb     = (const float*)d_in[10];
  const float* Alogs   = (const float*)d_in[11];
  const float* Dsv     = (const float*)d_in[12];
  const float* lnw     = (const float*)d_in[13];
  const float* lnb     = (const float*)d_in[14];
  const float* outw[2] = {(const float*)d_in[15], (const float*)d_in[16]};
  float* out = (float*)d_out;

  char* p = (char*)d_ws;
  auto alloc = [&](size_t bytes) -> char* {
    char* r = p; p += (bytes + 255) & ~(size_t)255; return r;
  };
  __bf16* win_bf   = (__bf16*)alloc((size_t)2 * 512 * 128 * sizeof(__bf16));
  __bf16* wout_bf  = (__bf16*)alloc((size_t)2 * 128 * 256 * sizeof(__bf16));
  __bf16* wproj_bf = (__bf16*)alloc((size_t)KDIR * NPROJ * DI * sizeof(__bf16));
  float*  Aneg     = (float*)alloc((size_t)KDIR * DI * NST * sizeof(float));
  __bf16* x_bf     = (__bf16*)alloc((size_t)M0 * 128 * sizeof(__bf16));
  float*  xz       = (float*)alloc((size_t)M0 * 512 * sizeof(float));
  float*  xc_f32   = (float*)alloc((size_t)M0 * 256 * sizeof(float));
  __bf16* xs0      = (__bf16*)alloc((size_t)M0 * 256 * sizeof(__bf16));
  __bf16* xs1      = (__bf16*)alloc((size_t)M0 * 256 * sizeof(__bf16));
  float*  xdbl     = (float*)alloc((size_t)KDIR * M0 * 64 * sizeof(float));
  float*  yo0      = (float*)alloc((size_t)M0 * 256 * sizeof(float));
  float*  yo1      = (float*)alloc((size_t)M0 * 256 * sizeof(float));
  __bf16* ybf      = (__bf16*)alloc((size_t)M0 * 256 * sizeof(__bf16));

  auto cvt = [&](const float* s, __bf16* dptr, int n) {
    k_f32_to_bf16<<<(n + 255) / 256, 256, 0, stream>>>(s, dptr, n);
  };
  cvt(inw[0], win_bf, 512 * 128);
  cvt(inw[1], win_bf + 512 * 128, 512 * 128);
  cvt(outw[0], wout_bf, 128 * 256);
  cvt(outw[1], wout_bf + 128 * 256, 128 * 256);
  k_pack_proj<<<(KDIR * NPROJ * DI + 255) / 256, 256, 0, stream>>>(xproj, wproj_bf);
  k_prep_A<<<(KDIR * DI * NST + 255) / 256, 256, 0, stream>>>(Alogs, Aneg, KDIR * DI * NST);

  for (int s = 0; s < 2; ++s) {
    // in_proj: (16384x128) x (512x128)^T -> (16384x512)
    cvt(xin[s], x_bf, M0 * 128);
    k_gemm_bf16<<<dim3(M0 / 128, 512 / 64), 256, 0, stream>>>(
        x_bf, win_bf + (size_t)s * 512 * 128, xz, M0, 512, 128);

    // depthwise conv + silu, emit scan input + both GEMM orderings
    k_conv_dw<<<B0 * L0, 256, 0, stream>>>(xz, cw[s], cb[s], xc_f32, xs0, xs1);

    // x_proj per direction: (16384x256) x (64x256)^T -> (16384x64), N padded
    for (int k = 0; k < KDIR; ++k)
      k_gemm_bf16<<<dim3(M0 / 128, 1), 256, 0, stream>>>(
          (k & 1) ? xs1 : xs0, wproj_bf + (size_t)k * NPROJ * DI,
          xdbl + (size_t)k * M0 * 64, M0, NPROJ, 256);

    // selective scan (16 blocks x 256 threads), forward+reverse accumulate
    k_zero<<<(M0 * 256 + 255) / 256, 256, 0, stream>>>(yo0, M0 * 256);
    k_zero<<<(M0 * 256 + 255) / 256, 256, 0, stream>>>(yo1, M0 * 256);
    k_scan<<<B0 * KDIR, 256, 0, stream>>>(xc_f32, xdbl, dtw, dtb, Aneg, Dsv, yo0, yo1);

    // cross-merge + LN + silu(z) gate -> bf16
    k_merge_ln<<<B0 * L0, 256, 0, stream>>>(yo0, yo1, xz, lnw, lnb, ybf);

    // out_proj: (16384x256) x (128x256)^T -> (16384x128) into d_out slice
    k_gemm_bf16<<<dim3(M0 / 128, 128 / 64), 256, 0, stream>>>(
        ybf, wout_bf + (size_t)s * 128 * 256, out + (size_t)s * M0 * 128,
        M0, 128, 256);
  }
}
